// Attension_66992899883449
// MI455X (gfx1250) — matrix operand order repaired and checked by reading, not compiled, not run on hardware
//
#include <hip/hip_runtime.h>
#include <hip/hip_fp16.h>

typedef __attribute__((ext_vector_type(16))) _Float16 v16h;
typedef __attribute__((ext_vector_type(8)))  float    v8f;

#define S_LEN   2048
#define D_DIM   128
#define BH_CNT  32          // B*H
#define BM      128         // query rows per workgroup (8 waves x 16)
#define BK      32          // keys per inner block
#define KS_PITCH 136        // halves: 128 + 8 pad (272B row, 16B aligned, bank-skewed)
#define VT_PITCH 40         // halves: 32 + 8 pad (80B row, 16B aligned)
#define NEGV    (-1.0e9f)

#if __has_builtin(__builtin_amdgcn_sched_barrier)
#define SCHED_FENCE() __builtin_amdgcn_sched_barrier(0)
#else
#define SCHED_FENCE()
#endif

union Frag { v16h v; uint4 q[2]; _Float16 h[16]; unsigned int u[8]; };
union H8   { uint4 q; _Float16 h[8]; };
union PkU  { unsigned int u[4]; _Float16 h[8]; };

// Cooperative staging of one 32-key K/V tile (f32 -> f16, V transposed).
__device__ __forceinline__ void stage_tile(_Float16* __restrict__ ksh,
                                           _Float16* __restrict__ vt,
                                           const float* __restrict__ Kb,
                                           const float* __restrict__ Vb,
                                           int j0, int tid)
{
    // K tile, row-major
    {
        const int k  = tid >> 3;
        const int d0 = (tid & 7) * 16;
        const float* src = Kb + (size_t)(j0 + k) * D_DIM + d0;
        H8 a, b;
        #pragma unroll
        for (int j = 0; j < 8; ++j) {
            a.h[j] = (_Float16)src[j];
            b.h[j] = (_Float16)src[8 + j];
        }
        *(uint4*)&ksh[k * KS_PITCH + d0]     = a.q;
        *(uint4*)&ksh[k * KS_PITCH + d0 + 8] = b.q;
    }
    // V tile, transposed: vt[d][k]
    {
        const int d  = tid & 127;
        const int kh = tid >> 7;
        #pragma unroll
        for (int kk = 0; kk < 2; ++kk) {
            const int k0 = kh * 8 + kk * 16;
            H8 t;
            #pragma unroll
            for (int i = 0; i < 8; ++i)
                t.h[i] = (_Float16)Vb[(size_t)(j0 + k0 + i) * D_DIM + d];
            *(uint4*)&vt[d * VT_PITCH + k0] = t.q;
        }
    }
}

__global__ __launch_bounds__(256) void fa_causal_fwd(
    const float* __restrict__ Q, const float* __restrict__ K,
    const float* __restrict__ V, const float* __restrict__ scale_p,
    float* __restrict__ O)
{
    // double-buffered tiles: 2 x (8.5 KB + 10 KB) = ~38 KB of the 320 KB WGP LDS
    __shared__ alignas(16) _Float16 Ksh[2][BK * KS_PITCH];
    __shared__ alignas(16) _Float16 Vt [2][D_DIM * VT_PITCH];

    const int tid  = threadIdx.x;
    const int lane = tid & 31;
    const int wv   = tid >> 5;
    const int g    = lane >> 4;   // lane group 0/1
    const int ln   = lane & 15;   // column index within 16

    const int qblk = blockIdx.x;
    const int bh   = blockIdx.y;
    const float scale = scale_p[0];

    const size_t base = (size_t)bh * S_LEN * D_DIM;
    const float* Qb = Q + base;
    const float* Kb = K + base;
    const float* Vb = V + base;
    float*       Ob = O + base;

    const int q0 = qblk * BM + wv * 16;   // first query row of this wave
    const int q  = q0 + ln;               // this lane's query column (softmax axis)

    // ---- preload Q^T as B-fragments: qb[c] half j = Q[q][32c + 16g + j] ----
    Frag qb[4];
    {
        const float* qr = Qb + (size_t)q * D_DIM;
        #pragma unroll
        for (int c = 0; c < 4; ++c) {
            const int d0 = 32 * c + 16 * g;
            #pragma unroll
            for (int j = 0; j < 16; ++j)
                qb[c].h[j] = (_Float16)qr[d0 + j];
        }
    }

    // O^T accumulators: 8 chunks of 16 d-rows; lane holds d = 16ch + 8g + r, col q
    v8f oacc[8];
    #pragma unroll
    for (int ch = 0; ch < 8; ++ch) {
        #pragma unroll
        for (int r = 0; r < 8; ++r) oacc[ch][r] = 0.0f;
    }
    float m_run = -1.0e30f;
    float l_run = 0.0f;

    const int nblk = (qblk + 1) * (BM / BK);

    // prologue: stage block 0 into buffer 0
    stage_tile(Ksh[0], Vt[0], Kb, Vb, 0, tid);

    for (int blk = 0; blk < nblk; ++blk) {
        const int j0  = blk * BK;
        const int cur = blk & 1;
        __syncthreads();   // staging of buffer `cur` complete; prior compute done

        // overlap: stage next block into the other buffer while computing this one
        if (blk + 1 < nblk) {
            stage_tile(Ksh[1 - cur], Vt[1 - cur], Kb, Vb, j0 + BK, tid);
            __builtin_prefetch(Kb + (size_t)(j0 + 2 * BK + (tid >> 3)) * D_DIM + (tid & 7) * 16, 0, 3);
            __builtin_prefetch(Vb + (size_t)(j0 + 2 * BK + (tid >> 3)) * D_DIM + (tid & 7) * 16, 0, 3);
        }

        if (j0 <= q0 + 15) {  // wave-uniform: at least one unmasked key in block
            const _Float16* ks = Ksh[cur];
            const _Float16* vs = Vt[cur];

            // ---- S^T = K * Q^T : two 16-key D tiles, contraction over d ----
            // Issue ALL 16 ds_load_b128 first into 8 live fragments; the
            // sched_barrier pins them above the WMMA chain so the waits become
            // partial (DS returns in order) and WMMAs overlap in-flight loads.
            Frag ka[8];
            #pragma unroll
            for (int c = 0; c < 4; ++c) {
                const int co = 32 * c + 16 * g;
                ka[2 * c    ].q[0] = *(const uint4*)&ks[ln * KS_PITCH + co];
                ka[2 * c    ].q[1] = *(const uint4*)&ks[ln * KS_PITCH + co + 8];
                ka[2 * c + 1].q[0] = *(const uint4*)&ks[(16 + ln) * KS_PITCH + co];
                ka[2 * c + 1].q[1] = *(const uint4*)&ks[(16 + ln) * KS_PITCH + co + 8];
            }
            SCHED_FENCE();

            v8f s0, s1;
            #pragma unroll
            for (int r = 0; r < 8; ++r) { s0[r] = 0.0f; s1[r] = 0.0f; }
            #pragma unroll
            for (int c = 0; c < 4; ++c) {
                s0 = __builtin_amdgcn_wmma_f32_16x16x32_f16(false, ka[2 * c].v, false, qb[c].v,
                                                            (short)0, s0, false, false);
                s1 = __builtin_amdgcn_wmma_f32_16x16x32_f16(false, ka[2 * c + 1].v, false, qb[c].v,
                                                            (short)0, s1, false, false);
            }
            SCHED_FENCE();

            // scale + causal mask: lane holds keys j0+8g+r (s0), j0+16+8g+r (s1)
            #pragma unroll
            for (int r = 0; r < 8; ++r) {
                const int kg = j0 + 8 * g + r;
                s0[r] = (kg      > q) ? NEGV : s0[r] * scale;
                s1[r] = (kg + 16 > q) ? NEGV : s1[r] * scale;
            }

            // ---- online softmax over this lane's query column ----
            float mx = NEGV;
            #pragma unroll
            for (int r = 0; r < 8; ++r) mx = fmaxf(mx, fmaxf(s0[r], s1[r]));
            mx = fmaxf(mx, __shfl_xor(mx, 16, 32));  // partner holds the other 16 keys
            const float m_new = fmaxf(m_run, mx);
            const float corr  = __expf(m_run - m_new);

            float p0[8], p1[8], sum = 0.0f;
            #pragma unroll
            for (int r = 0; r < 8; ++r) {
                p0[r] = __expf(s0[r] - m_new);
                p1[r] = __expf(s1[r] - m_new);
                sum += p0[r] + p1[r];
            }
            sum += __shfl_xor(sum, 16, 32);
            l_run = l_run * corr + sum;
            m_run = m_new;

            // ---- repack P into B-fragment (halves j = key 16g + j) ----
            PkU pk0, pk1;
            #pragma unroll
            for (int r = 0; r < 8; ++r) {
                pk0.h[r] = (_Float16)p0[r];
                pk1.h[r] = (_Float16)p1[r];
            }
            unsigned int x0[4], x1[4];
            #pragma unroll
            for (int i = 0; i < 4; ++i) {
                x0[i] = (unsigned int)__shfl_xor((int)pk0.u[i], 16, 32);
                x1[i] = (unsigned int)__shfl_xor((int)pk1.u[i], 16, 32);
            }
            Frag pb;
            #pragma unroll
            for (int i = 0; i < 4; ++i) {
                pb.u[i]     = g ? x1[i]    : pk0.u[i];  // keys 16g+0..7
                pb.u[4 + i] = g ? pk1.u[i] : x0[i];     // keys 16g+8..15
            }

            // ---- O^T += V^T * P : 8 d-chunks ----
            // all 16 V-fragment loads + the O rescale run above the fence,
            // the WMMA chain consumes below it with partial dscnt waits
            Frag va[8];
            #pragma unroll
            for (int ch = 0; ch < 8; ++ch) {
                const int d = 16 * ch + ln;
                va[ch].q[0] = *(const uint4*)&vs[d * VT_PITCH + 16 * g];
                va[ch].q[1] = *(const uint4*)&vs[d * VT_PITCH + 16 * g + 8];
            }
            #pragma unroll
            for (int ch = 0; ch < 8; ++ch) oacc[ch] *= corr;
            SCHED_FENCE();
            #pragma unroll
            for (int ch = 0; ch < 8; ++ch) {
                oacc[ch] = __builtin_amdgcn_wmma_f32_16x16x32_f16(false, va[ch].v, false, pb.v,
                                                                  (short)0, oacc[ch], false, false);
            }
            SCHED_FENCE();
        }
    }

    // ---- epilogue: divide by row sum (lane-local) and store ----
    const float inv_l = 1.0f / l_run;
    #pragma unroll
    for (int ch = 0; ch < 8; ++ch) {
        float* dst = Ob + (size_t)q * D_DIM + 16 * ch + 8 * g;
        float4 lo, hi;
        lo.x = oacc[ch][0] * inv_l; lo.y = oacc[ch][1] * inv_l;
        lo.z = oacc[ch][2] * inv_l; lo.w = oacc[ch][3] * inv_l;
        hi.x = oacc[ch][4] * inv_l; hi.y = oacc[ch][5] * inv_l;
        hi.z = oacc[ch][6] * inv_l; hi.w = oacc[ch][7] * inv_l;
        *(float4*)(dst)     = lo;
        *(float4*)(dst + 4) = hi;
    }
}

extern "C" void kernel_launch(void* const* d_in, const int* in_sizes, int n_in,
                              void* d_out, int out_size, void* d_ws, size_t ws_size,
                              hipStream_t stream) {
    (void)in_sizes; (void)n_in; (void)out_size; (void)d_ws; (void)ws_size;
    const float* Q = (const float*)d_in[0];
    const float* K = (const float*)d_in[1];
    const float* V = (const float*)d_in[2];
    const float* s = (const float*)d_in[3];
    float* O = (float*)d_out;

    dim3 grid(S_LEN / BM, BH_CNT);   // 16 x 32 workgroups
    dim3 block(256);                 // 8 waves (wave32)
    fa_causal_fwd<<<grid, block, 0, stream>>>(Q, K, V, s, O);
}
